// PhaseCodebook_57088705298853
// MI455X (gfx1250) — compile-verified, matrix operand order checked by reading, not run on hardware
//
#include <hip/hip_runtime.h>
#include <hip/hip_bf16.h>

// Problem constants (from the reference)
#define BDIM   8
#define SDIM   2048
#define NTOK   (BDIM * SDIM)   // 16384 tokens
#define VDIM   1024
#define EDIM   256
#define KCODES 8192

typedef __attribute__((ext_vector_type(16))) __bf16 v16bf;
typedef __attribute__((ext_vector_type(8)))  __bf16 v8bf;
typedef __attribute__((ext_vector_type(8)))  float  v8f;

// ---------------------------------------------------------------------------
// WMMA fragment loader, per CDNA5 ISA 7.12.2 16-bit A/B layouts (wave32):
//   lanes 0-15  : row/col = l, K = k0+0..7  and k0+16..23
//   lanes 16-31 : row/col = l, K = k0+8..15 and k0+24..31
// Two contiguous 16-byte chunks -> 2x (global_load_b128 | ds_load_b128).
// Works for generic pointers into either global memory or LDS.
// ---------------------------------------------------------------------------
__device__ __forceinline__ v16bf load_frag(const __bf16* p0, int row, int ldm,
                                           int k0, int half) {
    const __bf16* p = p0 + (size_t)row * ldm + k0 + 8 * half;
    v8bf lo = *(const v8bf*)(p);
    v8bf hi = *(const v8bf*)(p + 16);
    return __builtin_shufflevector(lo, hi, 0, 1, 2, 3, 4, 5, 6, 7,
                                           8, 9, 10, 11, 12, 13, 14, 15);
}

// ---------------------------------------------------------------------------
// Prep A: W (f32, [EDIM,VDIM]) -> bf16 copy (L2-resident operand of GEMM1).
// ---------------------------------------------------------------------------
__global__ void __launch_bounds__(256)
vq_wprep_kernel(const float* __restrict__ W, __bf16* __restrict__ W_bf16) {
    const size_t i0 = ((size_t)blockIdx.x * 256 + threadIdx.x) * 4;
#pragma unroll
    for (int j = 0; j < 4; ++j) W_bf16[i0 + j] = (__bf16)W[i0 + j];
}

// ---------------------------------------------------------------------------
// Prep B: per-code squared norm + bf16 copy of the codebook.
// ---------------------------------------------------------------------------
__global__ void __launch_bounds__(64)
vq_cbprep_kernel(const float* __restrict__ codebook, float* __restrict__ cnorm,
                 __bf16* __restrict__ cb_bf16) {
    const int k = blockIdx.x;
    const int t = threadIdx.x;
    const float* row = codebook + (size_t)k * EDIM;
    float s = 0.f;
    for (int j = t; j < EDIM; j += 64) {
        const float v = row[j];
        s += v * v;
        cb_bf16[(size_t)k * EDIM + j] = (__bf16)v;
    }
    __shared__ float red[64];
    red[t] = s;
    __syncthreads();
    for (int off = 32; off > 0; off >>= 1) {
        if (t < off) red[t] += red[t + off];
        __syncthreads();
    }
    if (t == 0) cnorm[k] = red[0];
}

// ---------------------------------------------------------------------------
// Kernel 1: projected = values @ W^T + bias   (WMMA bf16, f32 accum)
// grid = NTOK/16 blocks, 512 threads (16 waves). The 16x1024 f32 token tile
// (one contiguous 64KB span) is converted to bf16 and staged in LDS ONCE per
// block; each wave then owns output cols [16w,16w+16) with A from ds_load and
// B (bf16 W) streamed from L2.
// ---------------------------------------------------------------------------
__global__ void __launch_bounds__(512)
vq_proj_kernel(const float* __restrict__ values, const __bf16* __restrict__ W_bf16,
               const float* __restrict__ bias, float* __restrict__ proj_f32,
               __bf16* __restrict__ proj_bf16) {
    const int m0   = blockIdx.x * 16;
    const int wave = threadIdx.x >> 5;   // 0..15
    const int lane = threadIdx.x & 31;
    const int half = lane >> 4;
    const int l    = lane & 15;
    const int col  = wave * 16 + l;      // output column == W row

    __shared__ __bf16 a_sm[16 * VDIM];   // 32 KB

    // Cooperative stage: 2048 v8 chunks, 512 threads * 4 chunks each.
    const float* src = values + (size_t)m0 * VDIM;
#pragma unroll
    for (int it = 0; it < 4; ++it) {
        const int c = threadIdx.x + it * 512;
        v8f f = *(const v8f*)(src + (size_t)c * 8);
        v8bf h;
#pragma unroll
        for (int i = 0; i < 8; ++i) h[i] = (__bf16)f[i];
        *(v8bf*)(&a_sm[c * 8]) = h;
    }
    __syncthreads();

    v8f c = {};
#pragma unroll 4
    for (int kf = 0; kf < VDIM / 32; ++kf) {
        const int k0 = kf * 32;
        v16bf a = load_frag(a_sm, l, VDIM, k0, half);          // ds_load_b128 x2
        v16bf b = load_frag(W_bf16, col, VDIM, k0, half);      // global_load_b128 x2
        c = __builtin_amdgcn_wmma_f32_16x16x32_bf16(false, a, false, b,
                                                    (short)0, c, false, false);
    }
    const float bv = bias[col];
#pragma unroll
    for (int i = 0; i < 8; ++i) {
        const int row = m0 + i + 8 * half;            // C layout: VGPR i -> M = i + 8*half
        const float v = c[i] + bv;
        proj_f32[(size_t)row * EDIM + col]  = v;
        proj_bf16[(size_t)row * EDIM + col] = (__bf16)v;
    }
}

// ---------------------------------------------------------------------------
// Kernel 2: fused distance GEMM + argmin.  argmin_k (|c_k|^2 - 2 p.c_k).
// grid = NTOK/64 blocks of 128 threads (4 waves). Each wave owns a 16-token
// tile (A resident in 64 VGPRs) and scans ALL 8192 codes. The 16-code B tile
// (8KB contiguous) is DMA'd global->LDS with async-to-LDS b128 (ASYNCcnt),
// double-buffered so the next tile's DMA overlaps this tile's 8 chained
// WMMAs fed by ds_load_b128. Codebook L2 traffic: 256 blocks * 4MB = 1GB
// (4x less than per-16-token streaming).
// ---------------------------------------------------------------------------
#define NCHUNK (KCODES / 16)   // 512

__device__ __forceinline__ void stage_async(const __bf16* gsrc, unsigned ldsOff,
                                            int t) {
    // 8KB tile: 128 threads x 4 x b128. ASYNCcnt += 4 per wave.
    const char* g = (const char*)gsrc;
#pragma unroll
    for (int it = 0; it < 4; ++it) {
        const unsigned off = (unsigned)(t + it * 128) * 16u;
        asm volatile("global_load_async_to_lds_b128 %0, %1, off"
                     :: "v"(ldsOff + off), "v"(g + off) : "memory");
    }
}

__global__ void __launch_bounds__(128)
vq_argmin_kernel(const __bf16* __restrict__ proj_bf16,
                 const __bf16* __restrict__ cb_bf16,
                 const float* __restrict__ cnorm, int* __restrict__ ids) {
    const int wave = threadIdx.x >> 5;   // 0..3
    const int lane = threadIdx.x & 31;
    const int half = lane >> 4;
    const int l    = lane & 15;
    const int t    = threadIdx.x;
    const int m0   = blockIdx.x * 64 + wave * 16;   // this wave's token tile

    __shared__ __bf16 b_sm[2][16 * EDIM];           // 2 x 8 KB double buffer
    const unsigned ldsBase0 = (unsigned)(size_t)(&b_sm[0][0]);
    const unsigned ldsBase1 = (unsigned)(size_t)(&b_sm[1][0]);

    // Preload the whole 16x256 A tile as 8 resident fragments (64 VGPRs).
    v16bf a[8];
#pragma unroll
    for (int kf = 0; kf < EDIM / 32; ++kf)
        a[kf] = load_frag(proj_bf16, m0 + l, EDIM, kf * 32, half);

    float best[8];
    int   bidx[8];
#pragma unroll
    for (int i = 0; i < 8; ++i) { best[i] = 3.4e38f; bidx[i] = 0x7fffffff; }

    // Prologue: DMA chunk 0 into buffer 0.
    stage_async(cb_bf16, ldsBase0, t);

    for (int c = 0; c < NCHUNK; ++c) {
        const int buf = c & 1;
        if (c + 1 < NCHUNK) {
            stage_async(cb_bf16 + (size_t)(c + 1) * 16 * EDIM,
                        buf ? ldsBase0 : ldsBase1, t);
            asm volatile("s_wait_asynccnt 0x4" ::: "memory");  // chunk c landed
        } else {
            asm volatile("s_wait_asynccnt 0x0" ::: "memory");
        }
        __syncthreads();                     // all waves' DMA shares visible

        const int n0  = c * 16;
        const int col = n0 + l;              // this lane's code id this chunk
        const float cn = cnorm[col];
        v8f acc = {};
        const __bf16* bsm = &b_sm[buf][0];
#pragma unroll
        for (int kf = 0; kf < EDIM / 32; ++kf) {
            v16bf b = load_frag(bsm, l, EDIM, kf * 32, half);  // ds_load_b128 x2
            acc = __builtin_amdgcn_wmma_f32_16x16x32_bf16(false, a[kf], false, b,
                                                          (short)0, acc, false, false);
        }
#pragma unroll
        for (int i = 0; i < 8; ++i) {
            const float s = cn - 2.0f * acc[i];   // |p|^2 constant in k: dropped
            if (s < best[i]) { best[i] = s; bidx[i] = col; }
        }
        __syncthreads();                     // done reading buf before overwrite
    }

    // Cross-lane argmin within each 16-lane half (xor masks < 16 stay in-half).
#pragma unroll
    for (int i = 0; i < 8; ++i) {
        float bm = best[i];
        int   bi = bidx[i];
#pragma unroll
        for (int off = 1; off < 16; off <<= 1) {
            const float om = __shfl_xor(bm, off, 32);
            const int   oi = __shfl_xor(bi, off, 32);
            if (om < bm || (om == bm && oi < bi)) { bm = om; bi = oi; }
        }
        best[i] = bm;
        bidx[i] = bi;
    }
    if (l == 0) {   // lane 0 -> tokens 0..7, lane 16 -> tokens 8..15
#pragma unroll
        for (int i = 0; i < 8; ++i) ids[m0 + i + 8 * half] = bidx[i];
    }
}

// ---------------------------------------------------------------------------
// Kernel 3: gather quantized rows, emit ids as float, per-token SSE to ws.
// ---------------------------------------------------------------------------
__global__ void __launch_bounds__(64)
vq_finalize_kernel(const int* __restrict__ ids, const float* __restrict__ codebook,
                   const float* __restrict__ proj_f32, float* __restrict__ out_q,
                   float* __restrict__ out_ids, float* __restrict__ sse) {
    const int n  = blockIdx.x;
    const int t  = threadIdx.x;
    const int id = ids[n];
    const float* q = codebook + (size_t)id * EDIM;
    const float* p = proj_f32 + (size_t)n * EDIM;
    float s = 0.f;
    for (int j = t; j < EDIM; j += 64) {
        const float qv = q[j];
        out_q[(size_t)n * EDIM + j] = qv;       // straight-through fwd == quantized
        const float d = p[j] - qv;
        s += d * d;
    }
    __shared__ float red[64];
    red[t] = s;
    __syncthreads();
    for (int off = 32; off > 0; off >>= 1) {
        if (t < off) red[t] += red[t + off];
        __syncthreads();
    }
    if (t == 0) {
        out_ids[n] = (float)id;
        sse[n] = red[0];
    }
}

// Deterministic fixed-order loss reduction (single block).
__global__ void __launch_bounds__(256)
vq_loss_kernel(const float* __restrict__ sse, float* __restrict__ out_loss) {
    const int t = threadIdx.x;
    float s = 0.f;
    for (int i = t; i < NTOK; i += 256) s += sse[i];
    __shared__ float red[256];
    red[t] = s;
    __syncthreads();
    for (int off = 128; off > 0; off >>= 1) {
        if (t < off) red[t] += red[t + off];
        __syncthreads();
    }
    if (t == 0)
        *out_loss = 1.25f * red[0] / ((float)NTOK * (float)EDIM);
}

// ---------------------------------------------------------------------------
extern "C" void kernel_launch(void* const* d_in, const int* in_sizes, int n_in,
                              void* d_out, int out_size, void* d_ws, size_t ws_size,
                              hipStream_t stream) {
    const float* values   = (const float*)d_in[0];   // [NTOK, VDIM]
    const float* W        = (const float*)d_in[1];   // [EDIM, VDIM]
    const float* bias     = (const float*)d_in[2];   // [EDIM]
    const float* codebook = (const float*)d_in[3];   // [KCODES, EDIM]

    // Workspace partition (256B-aligned slices)
    char* ws = (char*)d_ws;
    size_t off = 0;
    auto take = [&](size_t bytes) {
        char* p = ws + off;
        off += (bytes + 255) & ~(size_t)255;
        return p;
    };
    float*  proj_f32  = (float*)take((size_t)NTOK * EDIM * sizeof(float));     // 16 MB
    __bf16* proj_bf16 = (__bf16*)take((size_t)NTOK * EDIM * sizeof(__bf16));   //  8 MB
    __bf16* cb_bf16   = (__bf16*)take((size_t)KCODES * EDIM * sizeof(__bf16)); //  4 MB
    __bf16* W_bf16    = (__bf16*)take((size_t)EDIM * VDIM * sizeof(__bf16));   // 512 KB
    float*  cnorm     = (float*)take((size_t)KCODES * sizeof(float));
    int*    ids       = (int*)take((size_t)NTOK * sizeof(int));
    float*  sse       = (float*)take((size_t)NTOK * sizeof(float));

    float* out_q    = (float*)d_out;                  // [NTOK, EDIM]
    float* out_ids  = out_q + (size_t)NTOK * EDIM;    // [NTOK]
    float* out_loss = out_ids + NTOK;                 // scalar

    vq_wprep_kernel<<<(EDIM * VDIM) / (256 * 4), 256, 0, stream>>>(W, W_bf16);
    vq_cbprep_kernel<<<KCODES, 64, 0, stream>>>(codebook, cnorm, cb_bf16);
    vq_proj_kernel<<<NTOK / 16, 512, 0, stream>>>(values, W_bf16, bias, proj_f32, proj_bf16);
    vq_argmin_kernel<<<NTOK / 64, 128, 0, stream>>>(proj_bf16, cb_bf16, cnorm, ids);
    vq_finalize_kernel<<<NTOK, 64, 0, stream>>>(ids, codebook, proj_f32, out_q, out_ids, sse);
    vq_loss_kernel<<<1, 256, 0, stream>>>(sse, out_loss);
}
